// SeqDecoderRNN_15341623182089
// MI455X (gfx1250) — matrix operand order repaired and checked by reading, not compiled, not run on hardware
//
#include <hip/hip_runtime.h>
#include <hip/hip_bf16.h>

// ---------------------------------------------------------------------------
// SeqDecoderRNN on MI455X (gfx1250, wave32, WMMA)
//
// 2-layer GRU scan, T=256 sequential steps. Weights are cast once to bf16
// (20 MB -> L2-resident on the 192 MB L2), and one persistent 64-WG kernel
// runs the recurrence with grid barriers through L2 atomics. Each WG owns a
// 16-wide hidden slice and computes all three gate blocks (r/z/n) for it, so
// the GRU elementwise update never leaves registers.
// Matrix math: v_wmma_f32_16x16x32_bf16 with f32 accumulation.
//
// Round-1: weight loads must not be hoisted out of the t-loop (round-0 asm
// showed LICM + scratch spill storms). Round-2 refinement: launder an integer
// OFFSET instead of the pointers themselves — the asm-opaque offset still
// blocks LICM, but the load addresses keep kernarg provenance so the
// infer-address-spaces pass selects global_load_b128 (round-1 asm had
// flat_load_b128, which ties up DScnt and the LDS arbitration path).
// ---------------------------------------------------------------------------

#define B_   64
#define H_   1024
#define C_   128
#define T_   256
#define NWG  64     // grid of persistent workgroups (must all be resident)
#define TPB  128    // 4 wave32 waves; wave i owns M-tile i (M = B = 64)

typedef __hip_bfloat16 bf16;
typedef __attribute__((ext_vector_type(16))) __bf16       v16bf;
typedef __attribute__((ext_vector_type(8)))  float        v8f;
typedef __attribute__((ext_vector_type(4))) unsigned int  v4u;

union Frag { v4u q[2]; v16bf v; };

// A-matrix 16x32 bf16 fragment (row-major source, leading dim ld).
// ISA layout: lane L holds row M=L%16; lane-half kh=L/16 holds
// K = kh*8 + {0..7} (VGPR0-3) and K = 16 + kh*8 + {0..7} (VGPR4-7).
__device__ __forceinline__ v16bf load_a_frag(const bf16* A, int ld, int m0, int k0) {
    const int lane = threadIdx.x & 31;
    const bf16* p  = A + (size_t)(m0 + (lane & 15)) * ld + k0 + ((lane >> 4) << 3);
    Frag f;
    f.q[0] = *reinterpret_cast<const v4u*>(p);
    f.q[1] = *reinterpret_cast<const v4u*>(p + 16);
    return f.v;
}

// B-matrix 32x16 bf16 fragment. For y = x @ W.T, B-tile column n is W row n
// (contiguous in K). ISA layout: lane L holds column N=L%16; lane-half kh
// holds K = kh*8 + {0..7} and K = 16 + kh*8 + {0..7}.
__device__ __forceinline__ v16bf load_b_frag(const bf16* W, int ld, int n0, int k0) {
    const int lane = threadIdx.x & 31;
    const bf16* p  = W + (size_t)(n0 + (lane & 15)) * ld + k0 + ((lane >> 4) << 3);
    Frag f;
    f.q[0] = *reinterpret_cast<const v4u*>(p);
    f.q[1] = *reinterpret_cast<const v4u*>(p + 16);
    return f.v;
}

__device__ __forceinline__ v8f wmma_bf16f32(v16bf a, v16bf b, v8f c) {
    // (neg_a, A, neg_b, B, c_mod, C, reuse_a, reuse_b)
    return __builtin_amdgcn_wmma_f32_16x16x32_bf16(false, a, false, b, (short)0, c,
                                                   false, false);
}

__device__ __forceinline__ float sigmoidf_(float x) {
    return 1.0f / (1.0f + __expf(-x));
}

// Grid-wide sense barrier over NWG resident workgroups (L2 atomics).
__device__ __forceinline__ void grid_barrier(unsigned* cnt, unsigned* gen) {
    __threadfence();            // make this WG's global stores device-visible
    __syncthreads();
    if (threadIdx.x == 0) {
        unsigned g = __hip_atomic_load(gen, __ATOMIC_RELAXED, __HIP_MEMORY_SCOPE_AGENT);
        unsigned a = __hip_atomic_fetch_add(cnt, 1u, __ATOMIC_ACQ_REL,
                                            __HIP_MEMORY_SCOPE_AGENT);
        if (a == NWG - 1) {
            __hip_atomic_store(cnt, 0u, __ATOMIC_RELAXED, __HIP_MEMORY_SCOPE_AGENT);
            __hip_atomic_fetch_add(gen, 1u, __ATOMIC_RELEASE, __HIP_MEMORY_SCOPE_AGENT);
        } else {
            while (__hip_atomic_load(gen, __ATOMIC_ACQUIRE,
                                     __HIP_MEMORY_SCOPE_AGENT) == g)
                __builtin_amdgcn_s_sleep(2);
        }
    }
    __syncthreads();
}

struct Gates { v8f r, z, n; };

// Accumulate the three 16x16 gate tiles for this wave's M-tile and the WG's
// hidden slice. W layout: (3H x K) row-major, gate g occupies rows [gH, gH+H).
__device__ __forceinline__ void accum_gates(Gates& g, const bf16* X, int K,
                                            const bf16* W, int mt, int hbase) {
    for (int k = 0; k < K; k += 32) {
        v16bf a = load_a_frag(X, K, mt * 16, k);
        g.r = wmma_bf16f32(a, load_b_frag(W, K, 0 * H_ + hbase, k), g.r);
        g.z = wmma_bf16f32(a, load_b_frag(W, K, 1 * H_ + hbase, k), g.z);
        g.n = wmma_bf16f32(a, load_b_frag(W, K, 2 * H_ + hbase, k), g.n);
    }
}

// GRU elementwise update done fully in registers on the accumulator layout:
// lane l, reg i -> (m = mt*16 + 8*(l>>4) + i, n = l&15).
__device__ __forceinline__ void gru_combine(const Gates& gi, const Gates& gh,
                                            float bir, float biz, float bin_,
                                            float bhr, float bhz, float bhn,
                                            float* hf, bf16* hbw, int mt, int col) {
    const int lane = threadIdx.x & 31;
    const int mb   = mt * 16 + ((lane >> 4) << 3);
#pragma unroll
    for (int i = 0; i < 8; ++i) {
        const int m = mb + i;
        float r  = sigmoidf_((gi.r[i] + bir) + (gh.r[i] + bhr));
        float z  = sigmoidf_((gi.z[i] + biz) + (gh.z[i] + bhz));
        float n  = tanhf((gi.n[i] + bin_) + r * (gh.n[i] + bhn));
        float hp = hf[(size_t)m * H_ + col];
        float hn = (1.0f - z) * n + z * hp;
        hf[(size_t)m * H_ + col]  = hn;
        hbw[(size_t)m * H_ + col] = __float2bfloat16(hn);
    }
}

__global__ __launch_bounds__(TPB, 1) void gru_decoder_persistent(
    const float* __restrict__ b_ih0, const float* __restrict__ b_hh0,
    const float* __restrict__ b_ih1, const float* __restrict__ b_hh1,
    const float* __restrict__ fc_b,
    const bf16* Wih0_, const bf16* Whh0_,
    const bf16* Wih1_, const bf16* Whh1_,
    const bf16* fcW_,
    float* h0f, float* h1f,
    bf16* h0p0, bf16* h0p1, bf16* h1p0, bf16* h1p1,
    bf16* xb, float* out, unsigned* bar)
{
    const int wg    = blockIdx.x;        // 0..63: owns hidden cols [16wg,16wg+16)
    const int mt    = threadIdx.x >> 5;  // wave id 0..3: owns batch rows [16mt,16mt+16)
    const int lane  = threadIdx.x & 31;
    const int hbase = wg * 16;
    const int col   = hbase + (lane & 15);

    // Per-lane gate biases (constant over the scan): hoist out of the t-loop.
    const float bir0 = b_ih0[col], biz0 = b_ih0[H_ + col], bin0 = b_ih0[2 * H_ + col];
    const float bhr0 = b_hh0[col], bhz0 = b_hh0[H_ + col], bhn0 = b_hh0[2 * H_ + col];
    const float bir1 = b_ih1[col], biz1 = b_ih1[H_ + col], bin1 = b_ih1[2 * H_ + col];
    const float bhr1 = b_hh1[col], bhz1 = b_hh1[H_ + col], bhn1 = b_hh1[2 * H_ + col];
    const float fcb  = (wg < 8) ? fc_b[wg * 16 + (lane & 15)] : 0.0f;

    unsigned* cnt = bar;
    unsigned* gen = bar + 1;

    // Opaque zero offset: redefined (as far as the compiler knows) every scan
    // step, so weight loads cannot be hoisted out of the t-loop, while the
    // addresses still root at the kernarg pointers -> global_load selection.
    size_t wofs = 0;

    for (int t = 0; t < T_; ++t) {
        asm volatile("" : "+s"(wofs));
        const bf16* Wih0 = Wih0_ + wofs;
        const bf16* Whh0 = Whh0_ + wofs;
        const bf16* Wih1 = Wih1_ + wofs;
        const bf16* Whh1 = Whh1_ + wofs;
        const bf16* fcW  = fcW_  + wofs;

        // ping-pong bf16 hidden-state buffers (cross-WG WMMA inputs)
        const bf16* h0r = (t & 1) ? h0p1 : h0p0;
        bf16*       h0w = (t & 1) ? h0p0 : h0p1;
        const bf16* h1r = (t & 1) ? h1p1 : h1p0;
        bf16*       h1w = (t & 1) ? h1p0 : h1p1;

        // -------- layer 0: gates from x (K=128) and h0 (K=1024) ----------
        {
            Gates gi = {}, gh = {};
            accum_gates(gi, xb,  C_, Wih0, mt, hbase);
            accum_gates(gh, h0r, H_, Whh0, mt, hbase);
            gru_combine(gi, gh, bir0, biz0, bin0, bhr0, bhz0, bhn0,
                        h0f, h0w, mt, col);
        }
        grid_barrier(cnt, gen);

        // -------- layer 1: gates from h0_new and h1 (both K=1024) --------
        {
            Gates gi = {}, gh = {};
            accum_gates(gi, h0w, H_, Wih1, mt, hbase);
            accum_gates(gh, h1r, H_, Whh1, mt, hbase);
            gru_combine(gi, gh, bir1, biz1, bin1, bhr1, bhz1, bhn1,
                        h1f, h1w, mt, col);
        }
        grid_barrier(cnt, gen);

        // -------- FC head + feedback: y = h1 @ fcW.T + fc_b --------------
        // 32 output tiles (4 M x 8 N) -> one tile per wave on WGs 0..7.
        if (wg < 8) {
            v8f acc = {};
            for (int k = 0; k < H_; k += 32)
                acc = wmma_bf16f32(load_a_frag(h1w, H_, mt * 16, k),
                                   load_b_frag(fcW, H_, wg * 16, k), acc);
            const int c  = wg * 16 + (lane & 15);
            const int mb = mt * 16 + ((lane >> 4) << 3);
#pragma unroll
            for (int i = 0; i < 8; ++i) {
                const int m = mb + i;
                float y = acc[i] + fcb;
                out[((size_t)m * C_ + c) * T_ + t] = y;   // (B, C, T) fp32
                xb[(size_t)m * C_ + c] = __float2bfloat16(y);
            }
        }
        grid_barrier(cnt, gen);
    }
}

// ------------------------------- prep kernels ------------------------------

__global__ void cvt_f32_bf16(const float* __restrict__ s, bf16* __restrict__ d, int n) {
    int i = blockIdx.x * blockDim.x + threadIdx.x;
    int stride = gridDim.x * blockDim.x;
    for (; i < n; i += stride) d[i] = __float2bfloat16(s[i]);
}

__global__ void init_state(const float* __restrict__ hidden,
                           const float* __restrict__ frame0,
                           float* h0f, float* h1f, bf16* h0b, bf16* h1b,
                           bf16* xb, unsigned* bar) {
    int i = blockIdx.x * blockDim.x + threadIdx.x;
    int stride = gridDim.x * blockDim.x;
    for (int j = i; j < B_ * H_; j += stride) {
        float a = hidden[j];             // layer 0 initial state
        float b = hidden[B_ * H_ + j];   // layer 1 initial state
        h0f[j] = a; h0b[j] = __float2bfloat16(a);
        h1f[j] = b; h1b[j] = __float2bfloat16(b);
    }
    for (int j = i; j < B_ * C_; j += stride)
        xb[j] = __float2bfloat16(frame0[j]);   // frame_0[:, :, 0]
    if (i < 8) bar[i] = 0u;   // workspace is poisoned -> re-arm barrier each call
}

// ------------------------------- launcher ----------------------------------

extern "C" void kernel_launch(void* const* d_in, const int* in_sizes, int n_in,
                              void* d_out, int out_size, void* d_ws, size_t ws_size,
                              hipStream_t stream) {
    (void)in_sizes; (void)n_in; (void)out_size; (void)ws_size;
    const float* hidden = (const float*)d_in[0];
    const float* frame0 = (const float*)d_in[1];
    const float* W_ih0  = (const float*)d_in[2];
    const float* W_hh0  = (const float*)d_in[3];
    const float* b_ih0  = (const float*)d_in[4];
    const float* b_hh0  = (const float*)d_in[5];
    const float* W_ih1  = (const float*)d_in[6];
    const float* W_hh1  = (const float*)d_in[7];
    const float* b_ih1  = (const float*)d_in[8];
    const float* b_hh1  = (const float*)d_in[9];
    const float* fc_W   = (const float*)d_in[10];
    const float* fc_b   = (const float*)d_in[11];
    float* out = (float*)d_out;

    char* ws = (char*)d_ws;
    size_t off = 0;
    auto alloc = [&](size_t bytes) -> char* {
        off = (off + 255) & ~size_t(255);
        char* p = ws + off;
        off += bytes;
        return p;
    };
    bf16*  Wih0b = (bf16*)alloc((size_t)3 * H_ * C_ * 2);
    bf16*  Whh0b = (bf16*)alloc((size_t)3 * H_ * H_ * 2);
    bf16*  Wih1b = (bf16*)alloc((size_t)3 * H_ * H_ * 2);
    bf16*  Whh1b = (bf16*)alloc((size_t)3 * H_ * H_ * 2);
    bf16*  fcWb  = (bf16*)alloc((size_t)C_ * H_ * 2);
    float* h0f   = (float*)alloc((size_t)B_ * H_ * 4);
    float* h1f   = (float*)alloc((size_t)B_ * H_ * 4);
    bf16*  h0p0  = (bf16*)alloc((size_t)B_ * H_ * 2);
    bf16*  h0p1  = (bf16*)alloc((size_t)B_ * H_ * 2);
    bf16*  h1p0  = (bf16*)alloc((size_t)B_ * H_ * 2);
    bf16*  h1p1  = (bf16*)alloc((size_t)B_ * H_ * 2);
    bf16*  xb    = (bf16*)alloc((size_t)B_ * C_ * 2);
    unsigned* bar = (unsigned*)alloc(256);

    // One-time (per call) fp32 -> bf16 weight cast; 20 MB stays L2-resident.
    cvt_f32_bf16<<<256, 256, 0, stream>>>(W_ih0, Wih0b, 3 * H_ * C_);
    cvt_f32_bf16<<<512, 256, 0, stream>>>(W_hh0, Whh0b, 3 * H_ * H_);
    cvt_f32_bf16<<<512, 256, 0, stream>>>(W_ih1, Wih1b, 3 * H_ * H_);
    cvt_f32_bf16<<<512, 256, 0, stream>>>(W_hh1, Whh1b, 3 * H_ * H_);
    cvt_f32_bf16<<<64,  256, 0, stream>>>(fc_W,  fcWb,  C_ * H_);
    init_state<<<256, 256, 0, stream>>>(hidden, frame0, h0f, h1f, h0p0, h1p0, xb, bar);

    // Persistent scan kernel: 64 WGs must be co-resident (trivial on MI455X).
    gru_decoder_persistent<<<NWG, TPB, 0, stream>>>(
        b_ih0, b_hh0, b_ih1, b_hh1, fc_b,
        Wih0b, Whh0b, Wih1b, Whh1b, fcWb,
        h0f, h1f, h0p0, h0p1, h1p0, h1p1, xb, out, bar);
}
